// GPSLayer_70196945486091
// MI455X (gfx1250) — matrix operand order repaired and checked by reading, not compiled, not run on hardware
//
#include <hip/hip_runtime.h>

#define N_NODES 4096
#define E_NUM   65536
#define D_MODEL 512
#define NEXP    4
#define DSTATE  16
#define DCONV   4
#define DTRANK  32

typedef __attribute__((ext_vector_type(16))) __bf16 bf16x16;
typedef __attribute__((ext_vector_type(8)))  float  f32x8;

__device__ __forceinline__ unsigned short f2bf(float f) {
  unsigned int u = __float_as_uint(f);
  u += 0x7FFFu + ((u >> 16) & 1u);           // round-to-nearest-even
  return (unsigned short)(u >> 16);
}
__device__ __forceinline__ float bf2f(unsigned short h) {
  return __uint_as_float(((unsigned int)h) << 16);
}
__device__ __forceinline__ float sigmoidf_(float x) { return 1.f / (1.f + __expf(-x)); }
__device__ __forceinline__ float softplusf_(float x) {
  return (x > 20.f) ? x : __logf(1.f + __expf(x));
}

// ---------------------------------------------------------------------------
// WMMA bf16 GEMM: C[M,Nout] = act(A[M,K] @ W[K,Nout] + bias)
// A: row-major bf16, staged via LDS (ds_load_b128 fragments).
// Bp: weights pre-swizzled to fragment order [K/32][Nout][32] so each lane's
//     fragment is two contiguous uint4 global loads (coalesced across lanes).
// Block: 256 threads (8 waves), tile 128x64, K step 32.
// act: 0=none 1=relu 2=softplus.  outBf16: write bf16 instead of f32.
// rowScale != null: C[row,col] += rowScale[row*rsStride] * val (f32 accumulate).
// ---------------------------------------------------------------------------
__global__ __launch_bounds__(256)
void k_gemm_bf16(const unsigned short* __restrict__ A,
                 const unsigned short* __restrict__ Bp,
                 void* __restrict__ Cout,
                 const float* __restrict__ bias,
                 const float* __restrict__ rowScale, int rsStride,
                 int K, int lda, int Nout, int ldc, int act, int outBf16)
{
  __shared__ unsigned short sA[128 * 32];
  const int wave = threadIdx.x >> 5;
  const int lane = threadIdx.x & 31;
  const int lm   = lane & 15;
  const int hi   = lane >> 4;
  const int bm   = blockIdx.y * 128;
  const int bn   = blockIdx.x * 64;

  f32x8 acc[4];
#pragma unroll
  for (int j = 0; j < 4; ++j)
#pragma unroll
    for (int r = 0; r < 8; ++r) acc[j][r] = 0.f;

  const unsigned int* Ag = (const unsigned int*)A;
  unsigned int* sAu = (unsigned int*)sA;

  for (int k0 = 0; k0 < K; k0 += 32) {
    // stage A tile 128x32 bf16 (2048 dwords, 8 per thread), coalesced
#pragma unroll
    for (int it = 0; it < 8; ++it) {
      int u = threadIdx.x + it * 256;
      int row = u >> 4, col2 = (u & 15) << 1;
      sAu[u] = Ag[(((size_t)(bm + row)) * lda + k0 + col2) >> 1];
    }
    __syncthreads();

    // A fragment: lane (lm,hi) reads two aligned 16B chunks of its row
    union { bf16x16 v; uint4 q[2]; } af;
    const unsigned short* arow = &sA[(wave * 16 + lm) * 32];
    af.q[0] = *(const uint4*)(arow + (hi << 4));
    af.q[1] = *(const uint4*)(arow + (hi << 4) + 8);

    // B fragments: two contiguous uint4 per lane from packed weights
    const unsigned short* bbase =
        Bp + (((size_t)(k0 >> 5)) * Nout + bn + lm) * 32 + (hi << 4);
#pragma unroll
    for (int j = 0; j < 4; ++j) {
      union { bf16x16 v; uint4 q[2]; } bfr;
      const uint4* bp = (const uint4*)(bbase + (size_t)j * 16 * 32);
      bfr.q[0] = bp[0];
      bfr.q[1] = bp[1];
      acc[j] = __builtin_amdgcn_wmma_f32_16x16x32_bf16(
          false, af.v, false, bfr.v, (short)0, acc[j], false, false);
    }
    __syncthreads();
  }

#pragma unroll
  for (int j = 0; j < 4; ++j) {
    int col = bn + j * 16 + lm;
    float bv = bias ? bias[col] : 0.f;
#pragma unroll
    for (int r = 0; r < 8; ++r) {
      int row = bm + wave * 16 + (hi << 3) + r;    // C/D layout: M = r + hi*8
      float v = acc[j][r] + bv;
      if (act == 1)      v = fmaxf(v, 0.f);
      else if (act == 2) v = softplusf_(v);
      size_t idx = (size_t)row * ldc + col;
      if (rowScale)      ((float*)Cout)[idx] += rowScale[(size_t)row * rsStride] * v;
      else if (outBf16)  ((unsigned short*)Cout)[idx] = f2bf(v);
      else               ((float*)Cout)[idx] = v;
    }
  }
}

// weight convert + swizzle to fragment order: dst[((k>>5)*Nout + n)*32 + (k&31)]
__global__ void k_cvt_swz(const float* __restrict__ src, unsigned short* __restrict__ dst,
                          int K, int Nout)
{
  long i = (long)blockIdx.x * blockDim.x + threadIdx.x;
  long total = (long)K * Nout;
  if (i >= total) return;
  int kk = (int)(i & 31);
  long t = i >> 5;
  int n  = (int)(t % Nout);
  int kb = (int)(t / Nout);
  dst[i] = f2bf(src[((size_t)kb * 32 + kk) * Nout + n]);
}

// --------------------------- GIN edge scatter ------------------------------
__global__ void k_scatter(const float* __restrict__ x, const int* __restrict__ ei,
                          const float* __restrict__ ea, float* __restrict__ aggr)
{
  long idx = (long)blockIdx.x * blockDim.x + threadIdx.x;
  long total = (long)E_NUM * D_MODEL;
  if (idx >= total) return;
  int e = (int)(idx / D_MODEL), d = (int)(idx % D_MODEL);
  int s = ei[e], dn = ei[E_NUM + e];
  float v = fmaxf(x[(size_t)s * D_MODEL + d] + ea[idx], 0.f);
  atomicAdd(&aggr[(size_t)dn * D_MODEL + d], v);
}

// --------------------------- elementwise helpers ---------------------------
__global__ void k_cvt(const float* __restrict__ s, unsigned short* __restrict__ d, long n) {
  long i = (long)blockIdx.x * blockDim.x + threadIdx.x;
  if (i < n) d[i] = f2bf(s[i]);
}
__global__ void k_addcvt(const float* __restrict__ a, const float* __restrict__ b,
                         unsigned short* __restrict__ d, long n) {
  long i = (long)blockIdx.x * blockDim.x + threadIdx.x;
  if (i < n) d[i] = f2bf(a[i] + b[i]);
}
__global__ void k_addstore(const float* __restrict__ a, const float* __restrict__ b,
                           float* __restrict__ h, unsigned short* __restrict__ hb, long n) {
  long i = (long)blockIdx.x * blockDim.x + threadIdx.x;
  if (i < n) { float v = a[i] + b[i]; h[i] = v; hb[i] = f2bf(v); }
}
__global__ void k_gz(const float* __restrict__ y, const float* __restrict__ xz,
                     unsigned short* __restrict__ g, long n) {
  long i = (long)blockIdx.x * blockDim.x + threadIdx.x;
  if (i >= n) return;
  int d = (int)(i % D_MODEL), t = (int)(i / D_MODEL);
  float z = xz[(size_t)t * (2 * D_MODEL) + D_MODEL + d];
  g[i] = f2bf(y[i] * z * sigmoidf_(z));
}

// depthwise causal conv (DCONV=4) + SiLU on xz[:, :D]
__global__ void k_conv(const float* __restrict__ xz, const float* __restrict__ cw,
                       const float* __restrict__ cb, float* __restrict__ xs,
                       unsigned short* __restrict__ xsb)
{
  long idx = (long)blockIdx.x * blockDim.x + threadIdx.x;
  long total = (long)N_NODES * D_MODEL;
  if (idx >= total) return;
  int t = (int)(idx / D_MODEL), d = (int)(idx % D_MODEL);
  float acc = cb[d];
#pragma unroll
  for (int k = 0; k < DCONV; ++k) {
    int tt = t - (DCONV - 1) + k;
    if (tt >= 0) acc += cw[d * DCONV + k] * xz[(size_t)tt * (2 * D_MODEL) + d];
  }
  float s = acc * sigmoidf_(acc);
  xs[idx] = s; xsb[idx] = f2bf(s);
}

// gate head: [N,256]bf16 @ [256,4] + softmax
__global__ void k_gate(const unsigned short* __restrict__ gh, const float* __restrict__ w2,
                       const float* __restrict__ b2, float* __restrict__ gate, int Nr)
{
  int n = blockIdx.x * blockDim.x + threadIdx.x;
  if (n >= Nr) return;
  float a0 = b2[0], a1 = b2[1], a2 = b2[2], a3 = b2[3];
  const unsigned short* r = gh + (size_t)n * 256;
  for (int k = 0; k < 256; ++k) {
    float v = bf2f(r[k]);
    a0 += v * w2[k * 4 + 0]; a1 += v * w2[k * 4 + 1];
    a2 += v * w2[k * 4 + 2]; a3 += v * w2[k * 4 + 3];
  }
  float m = fmaxf(fmaxf(a0, a1), fmaxf(a2, a3));
  float e0 = __expf(a0 - m), e1 = __expf(a1 - m), e2 = __expf(a2 - m), e3 = __expf(a3 - m);
  float s = e0 + e1 + e2 + e3;
  gate[n * 4 + 0] = e0 / s; gate[n * 4 + 1] = e1 / s;
  gate[n * 4 + 2] = e2 / s; gate[n * 4 + 3] = e3 / s;
}

// Mamba selective scan: lane quads own 4 states of one channel; B/C tiled in LDS.
__global__ __launch_bounds__(256)
void k_scan(const float* __restrict__ dt, const float* __restrict__ dbc,
            const float* __restrict__ xs, const float* __restrict__ A_log,
            const float* __restrict__ Dp, float* __restrict__ y, int L)
{
  __shared__ float sB[128][DSTATE];
  __shared__ float sC[128][DSTATE];
  int gt = blockIdx.x * 256 + threadIdx.x;
  int d = gt >> 2, q = gt & 3, s0 = q * 4;
  float Ai[4], hst[4] = {0.f, 0.f, 0.f, 0.f};
#pragma unroll
  for (int i = 0; i < 4; ++i) Ai[i] = -__expf(A_log[d * DSTATE + s0 + i]);
  float Dv = Dp[d];
  for (int T0 = 0; T0 < L; T0 += 128) {
    __syncthreads();
#pragma unroll
    for (int it = 0; it < 16; ++it) {              // 128*32 floats / 256 threads
      int idx = threadIdx.x + it * 256;
      int tt = idx >> 5, c = idx & 31;
      float v = dbc[(size_t)(T0 + tt) * 64 + DTRANK + c];
      if (c < DSTATE) sB[tt][c] = v; else sC[tt][c - DSTATE] = v;
    }
    __syncthreads();
    for (int tt = 0; tt < 128; ++tt) {
      int t = T0 + tt;
      float dtv = dt[(size_t)t * D_MODEL + d];
      float xv  = xs[(size_t)t * D_MODEL + d];
      float part = 0.f;
#pragma unroll
      for (int i = 0; i < 4; ++i) {
        float dA = __expf(dtv * Ai[i]);
        hst[i] = dA * hst[i] + dtv * sB[tt][s0 + i] * xv;
        part += hst[i] * sC[tt][s0 + i];
      }
      part += __shfl_xor(part, 1, 32);
      part += __shfl_xor(part, 2, 32);
      if (q == 0) y[(size_t)t * D_MODEL + d] = part + Dv * xv;
    }
  }
}

// batch-norm over rows: out = ((a[+b]) - mu)*rsqrt(var+eps)*g + beta, one block/col
__global__ void k_bn(const float* __restrict__ a, const float* __restrict__ b,
                     const float* __restrict__ g, const float* __restrict__ bet,
                     float* __restrict__ out, int Nr, int D)
{
  int c = blockIdx.x, t = threadIdx.x;
  float s = 0.f, q = 0.f;
  for (int n = t; n < Nr; n += 256) {
    float v = a[(size_t)n * D + c]; if (b) v += b[(size_t)n * D + c];
    s += v; q += v * v;
  }
  __shared__ float rs[256], rq[256];
  rs[t] = s; rq[t] = q; __syncthreads();
  for (int o = 128; o > 0; o >>= 1) {
    if (t < o) { rs[t] += rs[t + o]; rq[t] += rq[t + o]; }
    __syncthreads();
  }
  __shared__ float mu, istd;
  if (t == 0) {
    float m = rs[0] / Nr; mu = m;
    istd = rsqrtf(rq[0] / Nr - m * m + 1e-5f);
  }
  __syncthreads();
  float gg = g[c], bb = bet[c];
  for (int n = t; n < Nr; n += 256) {
    float v = a[(size_t)n * D + c]; if (b) v += b[(size_t)n * D + c];
    out[(size_t)n * D + c] = (v - mu) * istd * gg + bb;
  }
}

static inline unsigned eblk(long n) { return (unsigned)((n + 255) / 256); }

extern "C" void kernel_launch(void* const* d_in, const int* in_sizes, int n_in,
                              void* d_out, int out_size, void* d_ws, size_t ws_size,
                              hipStream_t stream)
{
  (void)in_sizes; (void)n_in; (void)out_size; (void)ws_size;
  const int N = N_NODES, D = D_MODEL;
  const float* x       = (const float*)d_in[0];
  const int*   ei      = (const int*)d_in[1];
  const float* ea      = (const float*)d_in[2];
  const float* gin_w1  = (const float*)d_in[3];
  const float* gin_b1  = (const float*)d_in[4];
  const float* gin_w2  = (const float*)d_in[5];
  const float* gin_b2  = (const float*)d_in[6];
  const float* bn1l_g  = (const float*)d_in[7];
  const float* bn1l_b  = (const float*)d_in[8];
  const float* bn1a_g  = (const float*)d_in[9];
  const float* bn1a_b  = (const float*)d_in[10];
  const float* bn2_g   = (const float*)d_in[11];
  const float* bn2_b   = (const float*)d_in[12];
  const float* gate_w1 = (const float*)d_in[13];
  const float* gate_b1 = (const float*)d_in[14];
  const float* gate_w2 = (const float*)d_in[15];
  const float* gate_b2 = (const float*)d_in[16];
  const float* in_w    = (const float*)d_in[17];
  const float* conv_w  = (const float*)d_in[18];
  const float* conv_b  = (const float*)d_in[19];
  const float* x_w     = (const float*)d_in[20];
  const float* dt_w    = (const float*)d_in[21];
  const float* dt_b    = (const float*)d_in[22];
  const float* A_log   = (const float*)d_in[23];
  const float* Dp      = (const float*)d_in[24];
  const float* out_w   = (const float*)d_in[25];
  const float* ff_w1   = (const float*)d_in[26];
  const float* ff_b1   = (const float*)d_in[27];
  const float* ff_w2   = (const float*)d_in[28];
  const float* ff_b2   = (const float*)d_in[29];
  float* out = (float*)d_out;

  char* ws = (char*)d_ws; size_t off = 0;
  auto alloc = [&](size_t bytes) -> void* {
    void* p = ws + off; off += (bytes + 255) & ~(size_t)255; return p;
  };
  float*          aggr  = (float*)alloc((size_t)N * D * 4);
  unsigned short* xb    = (unsigned short*)alloc((size_t)N * D * 2);
  unsigned short* a1b   = (unsigned short*)alloc((size_t)N * D * 2);
  unsigned short* t1b   = (unsigned short*)alloc((size_t)N * D * 2);
  unsigned short* wb    = (unsigned short*)alloc((size_t)D * 2 * D * 2);  // max 512x1024 bf16
  float*          hloc  = (float*)alloc((size_t)N * D * 4);
  unsigned short* gh    = (unsigned short*)alloc((size_t)N * 256 * 2);
  float*          gate  = (float*)alloc((size_t)N * 4 * 4);
  float*          xz    = (float*)alloc((size_t)N * 2 * D * 4);
  float*          xs    = (float*)alloc((size_t)N * D * 4);
  unsigned short* xsb   = (unsigned short*)alloc((size_t)N * D * 2);
  float*          dbc   = (float*)alloc((size_t)N * 64 * 4);
  unsigned short* dbcb  = (unsigned short*)alloc((size_t)N * 64 * 2);
  float*          dtb   = (float*)alloc((size_t)N * D * 4);
  float*          yscan = (float*)alloc((size_t)N * D * 4);
  unsigned short* gb    = (unsigned short*)alloc((size_t)N * D * 2);
  float*          hattn = (float*)alloc((size_t)N * D * 4);
  float*          h     = (float*)alloc((size_t)N * D * 4);
  unsigned short* hb    = (unsigned short*)alloc((size_t)N * D * 2);
  unsigned short* ff1b  = (unsigned short*)alloc((size_t)N * 2 * D * 2);
  float*          ffout = (float*)alloc((size_t)N * D * 4);

  auto gemm = [&](const unsigned short* A, const unsigned short* Bw, void* C,
                  const float* bias, const float* rs, int rsStride,
                  int M, int K, int Nout, int lda, int ldc, int act, int obf) {
    dim3 g(Nout / 64, M / 128);
    k_gemm_bf16<<<g, 256, 0, stream>>>(A, Bw, C, bias, rs, rsStride,
                                       K, lda, Nout, ldc, act, obf);
  };
  auto cvtw = [&](const float* src, int K, int Nout) {
    k_cvt_swz<<<eblk((long)K * Nout), 256, 0, stream>>>(src, wb, K, Nout);
  };

  // ---- GIN local branch ----
  hipMemsetAsync(aggr, 0, (size_t)N * D * 4, stream);
  k_scatter<<<eblk((long)E_NUM * D), 256, 0, stream>>>(x, ei, ea, aggr);
  k_cvt<<<eblk((long)N * D), 256, 0, stream>>>(x, xb, (long)N * D);
  k_addcvt<<<eblk((long)N * D), 256, 0, stream>>>(x, aggr, a1b, (long)N * D);
  cvtw(gin_w1, D, D);
  gemm(a1b, wb, t1b, gin_b1, nullptr, 0, N, D, D, D, D, 1, 1);
  cvtw(gin_w2, D, D);
  gemm(t1b, wb, hloc, gin_b2, nullptr, 0, N, D, D, D, D, 0, 0);
  k_bn<<<D, 256, 0, stream>>>(x, hloc, bn1l_g, bn1l_b, hloc, N, D);

  // ---- gate ----
  cvtw(gate_w1, D, 256);
  gemm(xb, wb, gh, gate_b1, nullptr, 0, N, D, 256, D, 256, 1, 1);
  k_gate<<<N / 256, 256, 0, stream>>>(gh, gate_w2, gate_b2, gate, N);

  // ---- Mamba experts (sequential, gate-weighted accumulate into hattn) ----
  hipMemsetAsync(hattn, 0, (size_t)N * D * 4, stream);
  for (int e = 0; e < NEXP; ++e) {
    cvtw(in_w + (size_t)e * D * 2 * D, D, 2 * D);
    gemm(xb, wb, xz, nullptr, nullptr, 0, N, D, 2 * D, D, 2 * D, 0, 0);
    k_conv<<<eblk((long)N * D), 256, 0, stream>>>(xz, conv_w + (size_t)e * D * DCONV,
                                                  conv_b + (size_t)e * D, xs, xsb);
    cvtw(x_w + (size_t)e * D * 64, D, 64);
    gemm(xsb, wb, dbc, nullptr, nullptr, 0, N, D, 64, D, 64, 0, 0);
    k_cvt<<<eblk((long)N * 64), 256, 0, stream>>>(dbc, dbcb, (long)N * 64);
    cvtw(dt_w + (size_t)e * DTRANK * D, DTRANK, D);
    gemm(dbcb, wb, dtb, dt_b + (size_t)e * D, nullptr, 0, N, DTRANK, D, 64, D, 2, 0);
    k_scan<<<8, 256, 0, stream>>>(dtb, dbc, xs, A_log + (size_t)e * D * DSTATE,
                                  Dp + (size_t)e * D, yscan, N);
    k_gz<<<eblk((long)N * D), 256, 0, stream>>>(yscan, xz, gb, (long)N * D);
    cvtw(out_w + (size_t)e * D * D, D, D);
    gemm(gb, wb, hattn, nullptr, gate + e, 4, N, D, D, D, D, 0, 0);
  }
  k_bn<<<D, 256, 0, stream>>>(x, hattn, bn1a_g, bn1a_b, hattn, N, D);

  // ---- combine + FFN + final BN ----
  k_addstore<<<eblk((long)N * D), 256, 0, stream>>>(hloc, hattn, h, hb, (long)N * D);
  cvtw(ff_w1, D, 2 * D);
  gemm(hb, wb, ff1b, ff_b1, nullptr, 0, N, D, 2 * D, D, 2 * D, 1, 1);
  cvtw(ff_w2, 2 * D, D);
  gemm(ff1b, wb, ffout, ff_b2, nullptr, 0, N, 2 * D, D, 2 * D, D, 0, 0);
  k_bn<<<D, 256, 0, stream>>>(h, ffout, bn2_g, bn2_b, out, N, D);
}
